// Value_20083267076605
// MI455X (gfx1250) — matrix operand order repaired and checked by reading, not compiled, not run on hardware
//
#include <hip/hip_runtime.h>
#include <math.h>

// ---------------- problem constants ----------------
#define S_    128
#define Dm    256         // D_OBS == D_ACT
#define Hh    8
#define DHh   32
#define Ll    4
#define Bb    4
#define MACT  (Bb*S_*S_)  // 65536 rows for the action stream
#define MOBS  (Bb*S_)     // 512 rows for the obs stream
#define NKEY  (S_+1)      // 129 keys per (b, s_obs)

typedef __attribute__((ext_vector_type(16))) __bf16 v16bf;
typedef __attribute__((ext_vector_type(8)))  float  v8f;

union FragBF { v16bf v; uint4 u[2]; unsigned short h[16]; };

static __device__ __forceinline__ unsigned short f2bf(float f) {
  unsigned int u = __float_as_uint(f);
  unsigned int r = u + 0x7FFFu + ((u >> 16) & 1u);   // round-nearest-even
  return (unsigned short)(r >> 16);
}
static __device__ __forceinline__ float bf2f(unsigned short h) {
  return __uint_as_float(((unsigned int)h) << 16);
}
static __device__ __forceinline__ float gelu_tanh(float x) {
  float x3 = x * x * x;
  return 0.5f * x * (1.0f + tanhf(0.7978845608028654f * (x + 0.044715f * x3)));
}
static __device__ __forceinline__ float wave_reduce_sum(float v) {
  #pragma unroll
  for (int m = 16; m >= 1; m >>= 1) v += __shfl_xor(v, m, 32);
  return v;
}

// ---------------- WMMA bf16 GEMM ----------------
// C(M,N) = A(M,K) * B(K,N) [+bias] [gelu]
// outMode: 1 = Cf32[row*N+col] += val ; 2 = Cbf16[orow*N+col] = val
// rowMode: 0 identity; 1 = act->concat-key remap ((r>>7)*129 + 1 + (r&127));
//          2 = obs->concat-key remap (r*129)
__global__ __launch_bounds__(256) void gemm_wmma_bf16(
    const unsigned short* __restrict__ A,
    const unsigned short* __restrict__ Bm,
    const float* __restrict__ bias,
    float* __restrict__ Cf,
    unsigned short* __restrict__ Cb,
    int M, int N, int K, int outMode, int rowMode, int doGelu)
{
  __shared__ __align__(16) unsigned short As [128][40];  // [m][k], padded
  __shared__ __align__(16) unsigned short BsT[128][40];  // [n][k], padded

  const int t     = threadIdx.x;
  const int lane  = t & 31;
  const int wid   = t >> 5;
  const int waveM = wid & 3;    // 4 waves along M
  const int waveN = wid >> 2;   // 2 waves along N
  const int halfl = lane >> 4;
  const int mBase = blockIdx.y * 128;
  const int nBase = blockIdx.x * 128;

  v8f acc[2][4];
  #pragma unroll
  for (int i = 0; i < 2; ++i)
    #pragma unroll
    for (int j = 0; j < 4; ++j)
      #pragma unroll
      for (int e = 0; e < 8; ++e) acc[i][j][e] = 0.0f;

  for (int k0 = 0; k0 < K; k0 += 32) {
    __syncthreads();
    { // A tile 128x32, row-major, 16 bf16 per thread
      int r = t >> 1, c = (t & 1) * 16;
      const unsigned short* src = A + (size_t)(mBase + r) * K + k0 + c;
      uint4 d0 = *(const uint4*)(src);
      uint4 d1 = *(const uint4*)(src + 8);
      *(uint4*)&As[r][c]     = d0;
      *(uint4*)&As[r][c + 8] = d1;
      if (k0 + 32 < K) __builtin_prefetch(src + 32, 0, 1);
    }
    { // B tile 32x128 -> transposed into BsT[n][k]
      int kr = t >> 3, c = (t & 7) * 16;
      const unsigned short* src = Bm + (size_t)(k0 + kr) * N + nBase + c;
      FragBF tmp;
      tmp.u[0] = *(const uint4*)(src);
      tmp.u[1] = *(const uint4*)(src + 8);
      #pragma unroll
      for (int j = 0; j < 16; ++j) BsT[c + j][kr] = tmp.h[j];
    }
    __syncthreads();

    // A fragment per CDNA5 layout: elems 0..7 -> K=halfl*8+0..7,
    // elems 8..15 -> K=16+halfl*8+0..7
    FragBF fa[2];
    #pragma unroll
    for (int mt = 0; mt < 2; ++mt) {
      int m = waveM * 32 + mt * 16 + (lane & 15);
      fa[mt].u[0] = *(const uint4*)&As[m][halfl * 8];
      fa[mt].u[1] = *(const uint4*)&As[m][16 + halfl * 8];
    }
    // B fragment: lane n = lane&15, elems 0..7 -> K=halfl*8+0..7, elems 8..15 -> K=16+halfl*8+0..7
    FragBF fb[4];
    #pragma unroll
    for (int nt = 0; nt < 4; ++nt) {
      int n = waveN * 64 + nt * 16 + (lane & 15);
      fb[nt].u[0] = *(const uint4*)&BsT[n][halfl * 8];
      fb[nt].u[1] = *(const uint4*)&BsT[n][16 + halfl * 8];
    }
    #pragma unroll
    for (int mt = 0; mt < 2; ++mt)
      #pragma unroll
      for (int nt = 0; nt < 4; ++nt)
        acc[mt][nt] = __builtin_amdgcn_wmma_f32_16x16x32_bf16(
            false, fa[mt].v, false, fb[nt].v, (short)0, acc[mt][nt], false, false);
  }

  // epilogue: C/D layout -> VGPR r holds row (r + 8*halfl), col = lane&15
  #pragma unroll
  for (int mt = 0; mt < 2; ++mt) {
    #pragma unroll
    for (int nt = 0; nt < 4; ++nt) {
      int col = nBase + waveN * 64 + nt * 16 + (lane & 15);
      float bv = bias ? bias[col] : 0.0f;
      #pragma unroll
      for (int r = 0; r < 8; ++r) {
        int row = mBase + waveM * 32 + mt * 16 + r + 8 * halfl;
        float val = acc[mt][nt][r] + bv;
        if (doGelu) val = gelu_tanh(val);
        size_t orow = (size_t)row;
        if (rowMode == 1)      orow = (size_t)(row >> 7) * NKEY + 1 + (row & (S_ - 1));
        else if (rowMode == 2) orow = (size_t)row * NKEY;
        size_t idx = orow * (size_t)N + col;
        if (outMode == 1) Cf[idx] += val;
        else              Cb[idx] = f2bf(val);
      }
    }
  }
}

// ---------------- LayerNorm (fp32 in -> bf16 out), D = 256 ----------------
__global__ __launch_bounds__(256) void layernorm_bf16(
    const float* __restrict__ x, const float* __restrict__ sc,
    const float* __restrict__ bi, unsigned short* __restrict__ y, int rows)
{
  int row  = blockIdx.x * 8 + (threadIdx.x >> 5);
  int lane = threadIdx.x & 31;
  if (row >= rows) return;
  const float* xr = x + (size_t)row * Dm + lane * 8;
  float v[8];
  float s = 0.0f, s2 = 0.0f;
  #pragma unroll
  for (int j = 0; j < 8; ++j) { v[j] = xr[j]; s += v[j]; s2 += v[j] * v[j]; }
  s  = wave_reduce_sum(s);
  s2 = wave_reduce_sum(s2);
  float mean = s * (1.0f / 256.0f);
  float var  = s2 * (1.0f / 256.0f) - mean * mean;
  float rstd = rsqrtf(var + 1e-6f);
  union { unsigned short h[8]; uint4 u; } o;
  #pragma unroll
  for (int j = 0; j < 8; ++j) {
    int c = lane * 8 + j;
    o.h[j] = f2bf((v[j] - mean) * rstd * sc[c] + bi[c]);
  }
  *(uint4*)(y + (size_t)row * Dm + lane * 8) = o.u;
}

// ---------------- WMMA flash attention -------------------------------------
// One block per (b, s_obs, head); 8 waves, each owns a 16-query tile.
// Q.K^T: DH=32 is one WMMA K step -> 9 WMMAs cover keys 0..143 (padded).
// Softmax across the 16-lane row group, P staged to LDS (zero padded to 160),
// V staged transposed; P.V = 10 WMMAs. Result added into act residual.
__global__ __launch_bounds__(256) void attention_wmma(
    const unsigned short* __restrict__ q,
    const unsigned short* __restrict__ kf,
    const unsigned short* __restrict__ vf,
    float* __restrict__ actres)
{
  __shared__ __align__(16) unsigned short Ks[144][32];     // [key][dim]
  __shared__ __align__(16) unsigned short VT[32][160];     // [dim][key], pad zeroed
  __shared__ __align__(16) unsigned short Pl[8][16][160];  // per-wave P tile

  const int t     = threadIdx.x;
  const int lane  = t & 31;
  const int wid   = t >> 5;
  const int halfl = lane >> 4;
  const int h  = blockIdx.x & 7;
  const int so = (blockIdx.x >> 3) & (S_ - 1);
  const int b  = blockIdx.x >> 10;
  const size_t rbase  = (size_t)(b * S_ + so) * S_;    // query row base
  const size_t kvbase = (size_t)(b * S_ + so) * NKEY;  // key/value row base

  // ---- stage K rows (129 x 32 bf16), zero pad rows 129..143 ----
  for (int c = t; c < 129 * 4; c += 256) {
    int key = c >> 2, qq = c & 3;
    *(uint4*)&Ks[key][qq * 8] =
        *(const uint4*)(kf + (kvbase + key) * Dm + h * DHh + qq * 8);
  }
  for (int c = t; c < 15 * 4; c += 256) {
    int key = 129 + (c >> 2), qq = c & 3;
    uint4 z; z.x = z.y = z.z = z.w = 0u;
    *(uint4*)&Ks[key][qq * 8] = z;
  }
  // ---- stage V transposed: VT[dim][key], zero pad keys 129..159 ----
  for (int key = t; key < 160; key += 256) {
    if (key < 129) {
      union { uint4 u[4]; unsigned short hh[32]; } tv;
      const unsigned short* vp = vf + (kvbase + key) * Dm + h * DHh;
      tv.u[0] = *(const uint4*)(vp);
      tv.u[1] = *(const uint4*)(vp + 8);
      tv.u[2] = *(const uint4*)(vp + 16);
      tv.u[3] = *(const uint4*)(vp + 24);
      #pragma unroll
      for (int d = 0; d < 32; ++d) VT[d][key] = tv.hh[d];
    } else {
      #pragma unroll
      for (int d = 0; d < 32; ++d) VT[d][key] = 0;
    }
  }
  __syncthreads();

  // ---- Q fragment straight from global (A layout) ----
  FragBF fq;
  {
    size_t qrow = rbase + wid * 16 + (lane & 15);
    const unsigned short* qp = q + qrow * Dm + h * DHh;
    fq.u[0] = *(const uint4*)(qp + halfl * 8);
    fq.u[1] = *(const uint4*)(qp + 16 + halfl * 8);
  }

  // ---- scores: 9 WMMAs over 16-key tiles ----
  v8f sc[9];
  #pragma unroll
  for (int kt = 0; kt < 9; ++kt) {
    #pragma unroll
    for (int e = 0; e < 8; ++e) sc[kt][e] = 0.0f;
    FragBF fk;
    int key = kt * 16 + (lane & 15);
    fk.u[0] = *(const uint4*)&Ks[key][halfl * 8];
    fk.u[1] = *(const uint4*)&Ks[key][16 + halfl * 8];
    sc[kt] = __builtin_amdgcn_wmma_f32_16x16x32_bf16(
        false, fq.v, false, fk.v, (short)0, sc[kt], false, false);
  }

  // ---- masked softmax per query row; write P (bf16) to LDS ----
  const float scale = 0.17677669529663687f;   // 1/sqrt(32)
  float lrow[8];
  #pragma unroll
  for (int r = 0; r < 8; ++r) {
    int qi = wid * 16 + r + 8 * halfl;        // query index in [0,128)
    float v[9];
    float mx = -3.0e38f;
    #pragma unroll
    for (int kt = 0; kt < 9; ++kt) {
      int kcol = kt * 16 + (lane & 15);
      bool vis = (kcol < NKEY) && (kcol <= qi + 1);   // obs key + causal actions
      v[kt] = vis ? sc[kt][r] * scale : -1.0e30f;
      mx = fmaxf(mx, v[kt]);
    }
    #pragma unroll
    for (int d = 8; d >= 1; d >>= 1) mx = fmaxf(mx, __shfl_xor(mx, d, 32));
    float sum = 0.0f;
    #pragma unroll
    for (int kt = 0; kt < 9; ++kt) { float p = __expf(v[kt] - mx); v[kt] = p; sum += p; }
    #pragma unroll
    for (int d = 8; d >= 1; d >>= 1) sum += __shfl_xor(sum, d, 32);
    lrow[r] = sum;
    int prow = r + 8 * halfl;
    #pragma unroll
    for (int kt = 0; kt < 9; ++kt)
      Pl[wid][prow][kt * 16 + (lane & 15)] = f2bf(v[kt]);
    Pl[wid][prow][144 + (lane & 15)] = 0;     // zero pad keys 144..159
  }

  // ---- P.V: 5 K-chunks x 2 dim tiles = 10 WMMAs ----
  v8f pv[2];
  #pragma unroll
  for (int nt = 0; nt < 2; ++nt)
    #pragma unroll
    for (int e = 0; e < 8; ++e) pv[nt][e] = 0.0f;
  #pragma unroll
  for (int kc = 0; kc < 5; ++kc) {
    FragBF fp;
    int m = lane & 15;
    fp.u[0] = *(const uint4*)&Pl[wid][m][kc * 32 + halfl * 8];
    fp.u[1] = *(const uint4*)&Pl[wid][m][kc * 32 + 16 + halfl * 8];
    #pragma unroll
    for (int nt = 0; nt < 2; ++nt) {
      int dim = nt * 16 + (lane & 15);
      FragBF fv;
      fv.u[0] = *(const uint4*)&VT[dim][kc * 32 + halfl * 8];
      fv.u[1] = *(const uint4*)&VT[dim][kc * 32 + 16 + halfl * 8];
      pv[nt] = __builtin_amdgcn_wmma_f32_16x16x32_bf16(
          false, fp.v, false, fv.v, (short)0, pv[nt], false, false);
    }
  }

  // ---- normalize by row sum and accumulate into residual ----
  #pragma unroll
  for (int nt = 0; nt < 2; ++nt) {
    int col = h * DHh + nt * 16 + (lane & 15);
    #pragma unroll
    for (int r = 0; r < 8; ++r) {
      int qi = wid * 16 + r + 8 * halfl;
      size_t idx = (rbase + qi) * Dm + col;
      actres[idx] += pv[nt][r] / lrow[r];
    }
  }
}

// ---------------- tiny helpers ----------------
__global__ void small_gemm_f32(const float* __restrict__ X, const float* __restrict__ W,
                               const float* __restrict__ bias, float* __restrict__ out,
                               int M, int N, int K)
{
  int idx = blockIdx.x * blockDim.x + threadIdx.x;
  if (idx >= M * N) return;
  int r = idx / N, c = idx % N;
  float s = bias[c];
  for (int k = 0; k < K; ++k) s += X[(size_t)r * K + k] * W[(size_t)k * N + c];
  out[idx] = s;
}

__global__ void act_resid_init(const float* __restrict__ ae, const float* __restrict__ pos,
                               float* __restrict__ out)
{
  int idx = blockIdx.x * 256 + threadIdx.x;    // over MACT*Dm
  int d = idx & 255;
  int r = idx >> 8;                            // r = (b*S+s)*S + q
  out[idx] = ae[(size_t)(r >> 7) * Dm + d] + pos[(size_t)(r & (S_ - 1)) * Dm + d];
}

__global__ void f32_to_bf16_k(const float* __restrict__ src, unsigned short* __restrict__ dst, int n)
{
  int i = blockIdx.x * 256 + threadIdx.x;
  if (i < n) dst[i] = f2bf(src[i]);
}

// ---------------- final LN + 256->1 projection ----------------
__global__ __launch_bounds__(256) void final_ln_proj(
    const float* __restrict__ x, const float* __restrict__ sc, const float* __restrict__ bi,
    const float* __restrict__ qw, const float* __restrict__ qb,
    float* __restrict__ out, int rows)
{
  int row  = blockIdx.x * 8 + (threadIdx.x >> 5);
  int lane = threadIdx.x & 31;
  if (row >= rows) return;
  const float* xr = x + (size_t)row * Dm + lane * 8;
  float v[8];
  float s = 0.0f, s2 = 0.0f;
  #pragma unroll
  for (int j = 0; j < 8; ++j) { v[j] = xr[j]; s += v[j]; s2 += v[j] * v[j]; }
  s  = wave_reduce_sum(s);
  s2 = wave_reduce_sum(s2);
  float mean = s * (1.0f / 256.0f);
  float var  = s2 * (1.0f / 256.0f) - mean * mean;
  float rstd = rsqrtf(var + 1e-6f);
  float dot = 0.0f;
  #pragma unroll
  for (int j = 0; j < 8; ++j) {
    int c = lane * 8 + j;
    float yv = (v[j] - mean) * rstd * sc[c] + bi[c];
    dot += yv * qw[c];
  }
  dot = wave_reduce_sum(dot);
  if (lane == 0) out[row] = dot + qb[0];
}

// ---------------- host-side orchestration ----------------
extern "C" void kernel_launch(void* const* d_in, const int* in_sizes, int n_in,
                              void* d_out, int out_size, void* d_ws, size_t ws_size,
                              hipStream_t stream) {
  const float* observations = (const float*)d_in[0];
  const float* actions      = (const float*)d_in[1];
  const float* act_embed_w  = (const float*)d_in[2];
  const float* act_embed_b  = (const float*)d_in[3];
  const float* pos_embed    = (const float*)d_in[4];
  const float* obs_embed_w  = (const float*)d_in[5];
  const float* obs_embed_b  = (const float*)d_in[6];
  const float* obs_ln1_s    = (const float*)d_in[7];
  const float* obs_ln1_b    = (const float*)d_in[8];
  const float* act_ln1_s    = (const float*)d_in[9];
  const float* act_ln1_b    = (const float*)d_in[10];
  const float* obs_ln2_s    = (const float*)d_in[11];
  const float* obs_ln2_b    = (const float*)d_in[12];
  const float* act_ln2_s    = (const float*)d_in[13];
  const float* act_ln2_b    = (const float*)d_in[14];
  const float* act_query_w  = (const float*)d_in[15];
  const float* act_query_b  = (const float*)d_in[16];
  const float* obs_key_w    = (const float*)d_in[17];
  const float* obs_key_b    = (const float*)d_in[18];
  const float* act_key_w    = (const float*)d_in[19];
  const float* act_key_b    = (const float*)d_in[20];
  const float* obs_value_w  = (const float*)d_in[21];
  const float* obs_value_b  = (const float*)d_in[22];
  const float* act_value_w  = (const float*)d_in[23];
  const float* act_value_b  = (const float*)d_in[24];
  const float* obs_mlp1_w   = (const float*)d_in[25];
  const float* obs_mlp1_b   = (const float*)d_in[26];
  const float* obs_mlp2_w   = (const float*)d_in[27];
  const float* obs_mlp2_b   = (const float*)d_in[28];
  const float* act_mlp1_w   = (const float*)d_in[29];
  const float* act_mlp1_b   = (const float*)d_in[30];
  const float* act_mlp2_w   = (const float*)d_in[31];
  const float* act_mlp2_b   = (const float*)d_in[32];
  const float* final_ln_s   = (const float*)d_in[33];
  const float* final_ln_b   = (const float*)d_in[34];
  const float* q_w          = (const float*)d_in[35];
  const float* q_b          = (const float*)d_in[36];

  char* ws = (char*)d_ws;
  size_t off = 0;
  auto alloc = [&](size_t bytes) -> char* {
    char* p = ws + off;
    off += (bytes + 255) & ~(size_t)255;
    return p;
  };

  float*          actres = (float*)alloc((size_t)MACT * Dm * 4);
  float*          obsres = (float*)alloc((size_t)MOBS * Dm * 4);
  float*          actemb = (float*)alloc((size_t)MOBS * Dm * 4);
  unsigned short* actn   = (unsigned short*)alloc((size_t)MACT * Dm * 2);
  unsigned short* obsn   = (unsigned short*)alloc((size_t)MOBS * Dm * 2);
  unsigned short* qbuf   = (unsigned short*)alloc((size_t)MACT * Dm * 2);
  unsigned short* kfull  = (unsigned short*)alloc((size_t)MOBS * NKEY * Dm * 2);
  unsigned short* vfull  = (unsigned short*)alloc((size_t)MOBS * NKEY * Dm * 2);
  unsigned short* hid    = (unsigned short*)alloc((size_t)MACT * 4 * Dm * 2);
  unsigned short* ohid   = (unsigned short*)alloc((size_t)MOBS * 4 * Dm * 2);
  unsigned short* wQ     = (unsigned short*)alloc((size_t)Ll * Dm * Dm * 2);
  unsigned short* wKo    = (unsigned short*)alloc((size_t)Ll * Dm * Dm * 2);
  unsigned short* wKa    = (unsigned short*)alloc((size_t)Ll * Dm * Dm * 2);
  unsigned short* wVo    = (unsigned short*)alloc((size_t)Ll * Dm * Dm * 2);
  unsigned short* wVa    = (unsigned short*)alloc((size_t)Ll * Dm * Dm * 2);
  unsigned short* wM1o   = (unsigned short*)alloc((size_t)Ll * Dm * 4 * Dm * 2);
  unsigned short* wM2o   = (unsigned short*)alloc((size_t)Ll * 4 * Dm * Dm * 2);
  unsigned short* wM1a   = (unsigned short*)alloc((size_t)Ll * Dm * 4 * Dm * 2);
  unsigned short* wM2a   = (unsigned short*)alloc((size_t)Ll * 4 * Dm * Dm * 2);

  auto conv = [&](const float* src, unsigned short* dst, int n) {
    f32_to_bf16_k<<<(n + 255) / 256, 256, 0, stream>>>(src, dst, n);
  };
  conv(act_query_w, wQ,  Ll * Dm * Dm);
  conv(obs_key_w,   wKo, Ll * Dm * Dm);
  conv(act_key_w,   wKa, Ll * Dm * Dm);
  conv(obs_value_w, wVo, Ll * Dm * Dm);
  conv(act_value_w, wVa, Ll * Dm * Dm);
  conv(obs_mlp1_w,  wM1o, Ll * Dm * 4 * Dm);
  conv(obs_mlp2_w,  wM2o, Ll * 4 * Dm * Dm);
  conv(act_mlp1_w,  wM1a, Ll * Dm * 4 * Dm);
  conv(act_mlp2_w,  wM2a, Ll * 4 * Dm * Dm);

  // embeddings
  small_gemm_f32<<<(MOBS * Dm + 255) / 256, 256, 0, stream>>>(
      actions, act_embed_w, act_embed_b, actemb, MOBS, Dm, 32);
  small_gemm_f32<<<(MOBS * Dm + 255) / 256, 256, 0, stream>>>(
      observations, obs_embed_w, obs_embed_b, obsres, MOBS, Dm, 128);
  act_resid_init<<<(MACT * Dm) / 256, 256, 0, stream>>>(actemb, pos_embed, actres);

  auto gemm = [&](const unsigned short* A, const unsigned short* Bm, const float* bias,
                  float* Cf, unsigned short* Cb, int M_, int N_, int K_,
                  int outMode, int rowMode, int doGelu) {
    dim3 grid(N_ / 128, M_ / 128);
    gemm_wmma_bf16<<<grid, 256, 0, stream>>>(A, Bm, bias, Cf, Cb, M_, N_, K_,
                                             outMode, rowMode, doGelu);
  };

  for (int i = 0; i < Ll; ++i) {
    // pre-attention layernorms
    layernorm_bf16<<<MOBS / 8, 256, 0, stream>>>(obsres, obs_ln1_s + i * Dm, obs_ln1_b + i * Dm, obsn, MOBS);
    layernorm_bf16<<<MACT / 8, 256, 0, stream>>>(actres, act_ln1_s + i * Dm, act_ln1_b + i * Dm, actn, MACT);
    // q / k / v projections (WMMA)
    gemm(actn, wQ  + (size_t)i * Dm * Dm, act_query_b + i * Dm, nullptr, qbuf,  MACT, Dm, Dm, 2, 0, 0);
    gemm(actn, wKa + (size_t)i * Dm * Dm, act_key_b   + i * Dm, nullptr, kfull, MACT, Dm, Dm, 2, 1, 0);
    gemm(actn, wVa + (size_t)i * Dm * Dm, act_value_b + i * Dm, nullptr, vfull, MACT, Dm, Dm, 2, 1, 0);
    gemm(obsn, wKo + (size_t)i * Dm * Dm, obs_key_b   + i * Dm, nullptr, kfull, MOBS, Dm, Dm, 2, 2, 0);
    gemm(obsn, wVo + (size_t)i * Dm * Dm, obs_value_b + i * Dm, nullptr, vfull, MOBS, Dm, Dm, 2, 2, 0);
    // WMMA flash attention, accumulate into act residual
    attention_wmma<<<Bb * S_ * Hh, 256, 0, stream>>>(qbuf, kfull, vfull, actres);
    // MLPs
    layernorm_bf16<<<MOBS / 8, 256, 0, stream>>>(obsres, obs_ln2_s + i * Dm, obs_ln2_b + i * Dm, obsn, MOBS);
    layernorm_bf16<<<MACT / 8, 256, 0, stream>>>(actres, act_ln2_s + i * Dm, act_ln2_b + i * Dm, actn, MACT);
    gemm(obsn, wM1o + (size_t)i * Dm * 4 * Dm, obs_mlp1_b + i * 4 * Dm, nullptr, ohid, MOBS, 4 * Dm, Dm, 2, 0, 1);
    gemm(ohid, wM2o + (size_t)i * 4 * Dm * Dm, obs_mlp2_b + i * Dm,     obsres, nullptr, MOBS, Dm, 4 * Dm, 1, 0, 0);
    gemm(actn, wM1a + (size_t)i * Dm * 4 * Dm, act_mlp1_b + i * 4 * Dm, nullptr, hid, MACT, 4 * Dm, Dm, 2, 0, 1);
    gemm(hid,  wM2a + (size_t)i * 4 * Dm * Dm, act_mlp2_b + i * Dm,     actres, nullptr, MACT, Dm, 4 * Dm, 1, 0, 0);
  }

  final_ln_proj<<<MACT / 8, 256, 0, stream>>>(actres, final_ln_s, final_ln_b,
                                              q_w, q_b, (float*)d_out, MACT);
}
